// MaskedMultiheadAttention_66700842107085
// MI455X (gfx1250) — matrix operand order repaired and checked by reading, not compiled, not run on hardware
//
#include <hip/hip_runtime.h>
#include <hip/hip_bf16.h>

// MI455X / gfx1250: wave32, WMMA f32_16x16x32_f16.
// B=4, S=2048, D=1024, H=16, DK=64.

typedef __attribute__((ext_vector_type(16))) _Float16 v16h;
typedef __attribute__((ext_vector_type(8)))  _Float16 v8h;
typedef __attribute__((ext_vector_type(8)))  float    v8f;

#define BATCH 4
#define SEQ   2048
#define DIM   1024
#define NHEAD 16
#define HDIM  64
#define NEGV  (-1.0e9f)

// ---- A fragment (16x32 f16, MxK). Row-major source, row = lane&15.
// lane<16: K {0..15}; lane>=16: K {16..31}.
__device__ __forceinline__ v16h load_frag_a(const _Float16* base, int ld) {
  const int lane = threadIdx.x & 31;
  const int r    = lane & 15;
  const int kh   = (lane >> 4) << 4;          // 0 or 16
  const _Float16* p = base + (size_t)r * ld + kh;
  v8h lo = *(const v8h*)(p);
  v8h hi = *(const v8h*)(p + 8);
  v16h f;
#pragma unroll
  for (int i = 0; i < 8; ++i) { f[i] = lo[i]; f[8 + i] = hi[i]; }
  return f;
}

// ---- B fragment (32x16 f16, KxN). base points at a [N][K] row-major view
// (row n = column n of B); lane n loads 16 consecutive K values starting at
// kb = (lane>=16 ? 16 : 0).
__device__ __forceinline__ v16h load_frag_b(const _Float16* base, int ld) {
  const int lane = threadIdx.x & 31;
  const int n    = lane & 15;
  const int kb   = (lane >> 4) << 4;          // 0 or 16
  const _Float16* p = base + (size_t)n * ld + kb;
  v8h lo = *(const v8h*)(p);
  v8h hi = *(const v8h*)(p + 8);
  v16h f;
#pragma unroll
  for (int i = 0; i < 8; ++i) { f[i] = lo[i]; f[8 + i] = hi[i]; }
  return f;
}

// Same, but lanes holding K>=16 return zeros when hi_ok==false (causal edge:
// avoids reading V^T columns past the valid sequence range).
__device__ __forceinline__ v16h load_frag_b_guard(const _Float16* base, int ld, bool hi_ok) {
  const int lane = threadIdx.x & 31;
  const int n    = lane & 15;
  const int kb   = (lane >> 4) << 4;
  v16h f;
  if (kb == 0 || hi_ok) {
    const _Float16* p = base + (size_t)n * ld + kb;
    v8h lo = *(const v8h*)(p);
    v8h hi = *(const v8h*)(p + 8);
#pragma unroll
    for (int i = 0; i < 8; ++i) { f[i] = lo[i]; f[8 + i] = hi[i]; }
  } else {
#pragma unroll
    for (int i = 0; i < 16; ++i) f[i] = (_Float16)0.0f;
  }
  return f;
}

// =====================================================================
// Kernel 1: y = x @ W^T + b  (f32 in, f16 out), out layout per head.
//   vtrans==0 : out[b][h][s][d]   (Q, K)
//   vtrans==1 : out[b][h][d][s]   (V transposed for contiguous A*V B-frags)
// Block: 256 thr (8 waves). Tile: 128(M) x 64(N), K step 32.
// =====================================================================
__global__ __launch_bounds__(256) void qkv_proj_kernel(
    const float* __restrict__ X, const float* __restrict__ W,
    const float* __restrict__ bias, _Float16* __restrict__ out, int vtrans) {
  __shared__ alignas(16) _Float16 sX[128 * 32];
  __shared__ alignas(16) _Float16 sW[64 * 32];
  const int tid  = threadIdx.x;
  const int wave = tid >> 5, lane = tid & 31;
  const int m0 = blockIdx.y * 128;
  const int n0 = blockIdx.x * 64;

  v8f acc[4];
#pragma unroll
  for (int nt = 0; nt < 4; ++nt)
#pragma unroll
    for (int j = 0; j < 8; ++j) acc[nt][j] = 0.0f;

  for (int k0 = 0; k0 < DIM; k0 += 32) {
    // stage X tile 128x32 (1024 float4 slots / 256 threads)
#pragma unroll
    for (int it = 0; it < 4; ++it) {
      const int slot = tid + it * 256;
      const int r = slot >> 3, c = (slot & 7) << 2;
      float4 xv = *(const float4*)(X + (size_t)(m0 + r) * DIM + k0 + c);
      _Float16* d = sX + r * 32 + c;
      d[0] = (_Float16)xv.x; d[1] = (_Float16)xv.y;
      d[2] = (_Float16)xv.z; d[3] = (_Float16)xv.w;
    }
    // stage W tile 64x32 (512 slots)
#pragma unroll
    for (int it = 0; it < 2; ++it) {
      const int slot = tid + it * 256;
      const int r = slot >> 3, c = (slot & 7) << 2;
      float4 wv = *(const float4*)(W + (size_t)(n0 + r) * DIM + k0 + c);
      _Float16* d = sW + r * 32 + c;
      d[0] = (_Float16)wv.x; d[1] = (_Float16)wv.y;
      d[2] = (_Float16)wv.z; d[3] = (_Float16)wv.w;
    }
    __syncthreads();
    v16h a = load_frag_a(sX + wave * 16 * 32, 32);
#pragma unroll
    for (int nt = 0; nt < 4; ++nt) {
      v16h bfr = load_frag_b(sW + nt * 16 * 32, 32);
      acc[nt] = __builtin_amdgcn_wmma_f32_16x16x32_f16(
          false, a, false, bfr, (short)0, acc[nt], false, false);
    }
    __syncthreads();
  }

  const int moff = (lane & 16) >> 1;  // 0 or 8
  const int nl   = lane & 15;
#pragma unroll
  for (int nt = 0; nt < 4; ++nt) {
    const int n  = n0 + nt * 16 + nl;
    const float bv = bias[n];
    const int h = n >> 6, d = n & 63;
#pragma unroll
    for (int j = 0; j < 8; ++j) {
      const int m = m0 + wave * 16 + moff + j;
      const int b = m >> 11, s = m & (SEQ - 1);
      const float v = acc[nt][j] + bv;
      const size_t idx = vtrans
          ? (((size_t)(b * NHEAD + h) * HDIM + d) * SEQ + s)
          : (((size_t)(b * NHEAD + h) * SEQ + s) * HDIM + d);
      out[idx] = (_Float16)v;
    }
  }
}

// =====================================================================
// Attention tile processor: NT 16-key subtiles (NT in 1..4) starting at k0.
// CAUSAL adds the per-element triangle test (only needed on the tail tile).
// One online-softmax update amortized over the whole macro-tile.
// =====================================================================
template <int NT, bool CAUSAL>
__device__ __forceinline__ void attn_tile(
    int k0, int qr0, const _Float16* __restrict__ kbase,
    const _Float16* __restrict__ vbase, const int* __restrict__ mb,
    _Float16* pw, const v16h& qf0, const v16h& qf1,
    float (&mrow)[8], float (&lrow)[8], v8f (&acc)[4]) {
  const int lane = threadIdx.x & 31;
  const int moff = (lane & 16) >> 1;
  const int nl   = lane & 15;

  // ---- scores: NT x (16x16) C-frags, K reduced over HDIM=64 in 2 WMMAs
  v8f sc[NT];
#pragma unroll
  for (int t = 0; t < NT; ++t) {
    const _Float16* kp = kbase + (size_t)(k0 + t * 16) * HDIM;
    v16h b0 = load_frag_b(kp, HDIM);
    v16h b1 = load_frag_b(kp + 32, HDIM);
    v8f z;
#pragma unroll
    for (int j = 0; j < 8; ++j) z[j] = 0.0f;
    sc[t] = __builtin_amdgcn_wmma_f32_16x16x32_f16(false, qf0, false, b0, (short)0, z,     false, false);
    sc[t] = __builtin_amdgcn_wmma_f32_16x16x32_f16(false, qf1, false, b1, (short)0, sc[t], false, false);
  }

  // ---- masks (C layout: col = nl + t*16, row = moff + j)
  int mk[NT];
#pragma unroll
  for (int t = 0; t < NT; ++t) mk[t] = mb[k0 + t * 16 + nl];
#pragma unroll
  for (int t = 0; t < NT; ++t) {
    const int c = k0 + t * 16 + nl;
#pragma unroll
    for (int j = 0; j < 8; ++j) {
      if (CAUSAL) {
        if ((c > qr0 + moff + j) || (mk[t] == 0)) sc[t][j] = NEGV;
      } else {
        if (mk[t] == 0) sc[t][j] = NEGV;
      }
    }
  }

  // ---- one online-softmax update for the whole macro-tile
#pragma unroll
  for (int j = 0; j < 8; ++j) {
    float tmax = sc[0][j];
#pragma unroll
    for (int t = 1; t < NT; ++t) tmax = fmaxf(tmax, sc[t][j]);
#pragma unroll
    for (int d2 = 1; d2 < 16; d2 <<= 1) tmax = fmaxf(tmax, __shfl_xor(tmax, d2, 32));
    const float mnew = fmaxf(mrow[j], tmax);
    const float corr = __expf(mrow[j] - mnew);
    mrow[j] = mnew;
    float p[NT], ps = 0.0f;
#pragma unroll
    for (int t = 0; t < NT; ++t) { p[t] = __expf(sc[t][j] - mnew); ps += p[t]; }
#pragma unroll
    for (int d2 = 1; d2 < 16; d2 <<= 1) ps += __shfl_xor(ps, d2, 32);
    lrow[j] = lrow[j] * corr + ps;
#pragma unroll
    for (int nt = 0; nt < 4; ++nt) acc[nt][j] *= corr;
    // C-layout -> per-wave LDS scratch (row stride 64)
#pragma unroll
    for (int t = 0; t < NT; ++t)
      pw[(moff + j) * 64 + t * 16 + nl] = (_Float16)p[t];
    if (NT & 1)  // zero the unused upper half of the last 32-wide A chunk
      pw[(moff + j) * 64 + NT * 16 + nl] = (_Float16)0.0f;
  }

  // ---- context += P (16 x NT*16) x V: ceil(NT/2) chunks of K=32
  constexpr int NCH = (NT + 1) / 2;
#pragma unroll
  for (int c = 0; c < NCH; ++c) {
    v16h pa = load_frag_a(pw + c * 32, 64);
    const bool hi_ok = (2 * c + 1) < NT;
#pragma unroll
    for (int nt = 0; nt < 4; ++nt) {
      v16h bv = load_frag_b_guard(
          vbase + (size_t)(nt * 16) * SEQ + k0 + c * 32, SEQ, hi_ok);
      acc[nt] = __builtin_amdgcn_wmma_f32_16x16x32_f16(
          false, pa, false, bv, (short)0, acc[nt], false, false);
    }
  }
}

// =====================================================================
// Kernel 2: causal flash attention. Grid (S/128, H, B), 256 thr = 8 waves.
// Each wave owns 16 query rows. Interior 64-key tiles skip the causal
// compare; the tail tile (1..4 subtiles) handles the triangle.
// =====================================================================
__global__ __launch_bounds__(256) void attn_kernel(
    const _Float16* __restrict__ Qh, const _Float16* __restrict__ Kh,
    const _Float16* __restrict__ Vt, const int* __restrict__ mask,
    _Float16* __restrict__ ctx) {
  __shared__ alignas(16) _Float16 sP[8 * 16 * 64];  // 16 KB
  const int tid  = threadIdx.x;
  const int wave = tid >> 5, lane = tid & 31;
  const int h = blockIdx.y, b = blockIdx.z;
  const int qr0 = blockIdx.x * 128 + wave * 16;
  const size_t hb = (size_t)(b * NHEAD + h) * (SEQ * HDIM);
  const int moff = (lane & 16) >> 1;
  const int nl   = lane & 15;

  const _Float16* kbase = Kh + hb;
  const _Float16* vbase = Vt + hb;
  const int* mb = mask + b * SEQ;
  _Float16* pw = sP + wave * (16 * 64);

  // Pre-load + pre-scale Q fragments (1/sqrt(64) = 0.125 is exact in f16).
  v16h qf0 = load_frag_a(Qh + hb + (size_t)qr0 * HDIM, HDIM);
  v16h qf1 = load_frag_a(Qh + hb + (size_t)qr0 * HDIM + 32, HDIM);
#pragma unroll
  for (int i = 0; i < 16; ++i) {
    qf0[i] = qf0[i] * (_Float16)0.125f;
    qf1[i] = qf1[i] * (_Float16)0.125f;
  }

  float mrow[8], lrow[8];
  v8f acc[4];
#pragma unroll
  for (int j = 0; j < 8; ++j) { mrow[j] = -3.0e38f; lrow[j] = 0.0f; }
#pragma unroll
  for (int nt = 0; nt < 4; ++nt)
#pragma unroll
    for (int j = 0; j < 8; ++j) acc[nt][j] = 0.0f;

  const int tail_start = qr0 & ~63;           // interior region [0, tail_start)
  const int ntail = ((qr0 + 16) - tail_start) >> 4;  // 1..4 subtiles

  for (int k0 = 0; k0 < tail_start; k0 += 64) {
    // prefetch next macro-tile of K (8 KB) and the V row-slices
    if (k0 + 64 < tail_start) {
      const _Float16* pk = kbase + (size_t)(k0 + 64) * HDIM + lane * 128;
      __builtin_prefetch(pk, 0, 3);
      __builtin_prefetch(pk + 64, 0, 3);
      __builtin_prefetch(vbase + (size_t)lane * SEQ + k0 + 64, 0, 3);
      __builtin_prefetch(vbase + (size_t)(lane + 32) * SEQ + k0 + 64, 0, 3);
    }
    attn_tile<4, false>(k0, qr0, kbase, vbase, mb, pw, qf0, qf1, mrow, lrow, acc);
  }
  switch (ntail) {
    case 1:  attn_tile<1, true>(tail_start, qr0, kbase, vbase, mb, pw, qf0, qf1, mrow, lrow, acc); break;
    case 2:  attn_tile<2, true>(tail_start, qr0, kbase, vbase, mb, pw, qf0, qf1, mrow, lrow, acc); break;
    case 3:  attn_tile<3, true>(tail_start, qr0, kbase, vbase, mb, pw, qf0, qf1, mrow, lrow, acc); break;
    default: attn_tile<4, true>(tail_start, qr0, kbase, vbase, mb, pw, qf0, qf1, mrow, lrow, acc); break;
  }

  // normalize + store ctx as f16 [B,S,D]
#pragma unroll
  for (int j = 0; j < 8; ++j) {
    const float inv = 1.0f / lrow[j];
    const int s = qr0 + moff + j;
#pragma unroll
    for (int nt = 0; nt < 4; ++nt) {
      ctx[((size_t)(b * SEQ + s)) * DIM + h * HDIM + nt * 16 + nl] =
          (_Float16)(acc[nt][j] * inv);
    }
  }
}

// =====================================================================
// Kernel 3: out = ctx @ Wp^T + bp (f16 ctx in, f32 out). Same tiling as k1;
// ctx is already f16 so A-frags load straight from global.
// =====================================================================
__global__ __launch_bounds__(256) void out_proj_kernel(
    const _Float16* __restrict__ ctx, const float* __restrict__ W,
    const float* __restrict__ bias, float* __restrict__ out) {
  __shared__ alignas(16) _Float16 sW[64 * 32];
  const int tid  = threadIdx.x;
  const int wave = tid >> 5, lane = tid & 31;
  const int m0 = blockIdx.y * 128;
  const int n0 = blockIdx.x * 64;

  v8f acc[4];
#pragma unroll
  for (int nt = 0; nt < 4; ++nt)
#pragma unroll
    for (int j = 0; j < 8; ++j) acc[nt][j] = 0.0f;

  for (int k0 = 0; k0 < DIM; k0 += 32) {
#pragma unroll
    for (int it = 0; it < 2; ++it) {
      const int slot = tid + it * 256;
      const int r = slot >> 3, c = (slot & 7) << 2;
      float4 wv = *(const float4*)(W + (size_t)(n0 + r) * DIM + k0 + c);
      _Float16* d = sW + r * 32 + c;
      d[0] = (_Float16)wv.x; d[1] = (_Float16)wv.y;
      d[2] = (_Float16)wv.z; d[3] = (_Float16)wv.w;
    }
    __syncthreads();
    v16h a = load_frag_a(ctx + (size_t)(m0 + wave * 16) * DIM + k0, DIM);
#pragma unroll
    for (int nt = 0; nt < 4; ++nt) {
      v16h bfr = load_frag_b(sW + nt * 16 * 32, 32);
      acc[nt] = __builtin_amdgcn_wmma_f32_16x16x32_f16(
          false, a, false, bfr, (short)0, acc[nt], false, false);
    }
    __syncthreads();
  }

  const int moff = (lane & 16) >> 1;
  const int nl   = lane & 15;
#pragma unroll
  for (int nt = 0; nt < 4; ++nt) {
    const int n = n0 + nt * 16 + nl;
    const float bv = bias[n];
#pragma unroll
    for (int j = 0; j < 8; ++j) {
      const int m = m0 + wave * 16 + moff + j;
      out[(size_t)m * DIM + n] = acc[nt][j] + bv;
    }
  }
}

extern "C" void kernel_launch(void* const* d_in, const int* in_sizes, int n_in,
                              void* d_out, int out_size, void* d_ws, size_t ws_size,
                              hipStream_t stream) {
  (void)in_sizes; (void)n_in; (void)out_size; (void)ws_size;
  const float* q    = (const float*)d_in[0];
  const float* k    = (const float*)d_in[1];
  const float* v    = (const float*)d_in[2];
  const int*   mask = (const int*)d_in[3];
  const float* Wq = (const float*)d_in[4];
  const float* bq = (const float*)d_in[5];
  const float* Wk = (const float*)d_in[6];
  const float* bk = (const float*)d_in[7];
  const float* Wv = (const float*)d_in[8];
  const float* bv = (const float*)d_in[9];
  const float* Wp = (const float*)d_in[10];
  const float* bp = (const float*)d_in[11];
  float* out = (float*)d_out;

  const size_t nelem = (size_t)BATCH * SEQ * DIM;
  _Float16* Qh  = (_Float16*)d_ws;
  _Float16* Kh  = Qh + nelem;
  _Float16* Vt  = Kh + nelem;
  _Float16* Ctx = Vt + nelem;

  const dim3 blk(256);
  const dim3 gp(DIM / 64, (BATCH * SEQ) / 128);           // 16 x 64
  qkv_proj_kernel<<<gp, blk, 0, stream>>>(q, Wq, bq, Qh, 0);
  qkv_proj_kernel<<<gp, blk, 0, stream>>>(k, Wk, bk, Kh, 0);
  qkv_proj_kernel<<<gp, blk, 0, stream>>>(v, Wv, bv, Vt, 1);
  attn_kernel<<<dim3(SEQ / 128, NHEAD, BATCH), blk, 0, stream>>>(Qh, Kh, Vt, mask, Ctx);
  out_proj_kernel<<<gp, blk, 0, stream>>>(Ctx, Wp, bp, out);
}
